// Generator_16776142258324
// MI455X (gfx1250) — matrix operand order repaired and checked by reading, not compiled, not run on hardware
//
#include <hip/hip_runtime.h>
#include <stdint.h>

typedef __attribute__((ext_vector_type(16))) __bf16 v16bf;
typedef __attribute__((ext_vector_type(8)))  __bf16 v8bf;
typedef __attribute__((ext_vector_type(8)))  float  v8f;

#define B_     1024
#define T_     128
#define IN_    32
#define H_     256
#define OUT_   32
#define NLAYER 4          // L+1 MLP layers in f

#define ROWS     16       // batch rows per workgroup
#define NTHREADS 256      // 8 waves (wave32); 32 async instrs/wave/layer (<= ASYNCcnt range)
#define NWAVES   8
#define AST      264      // padded bf16 row stride (H_ + 8) -> bank-conflict-free ds_load_b128

// ---- dynamic LDS layout (bytes) ----
#define OFF_W0    0
#define SZ_W      (H_ * AST * 2)        // 135168: one layer of W^T in bf16
#define OFF_W1    (OFF_W0 + SZ_W)
#define OFF_A0    (OFF_W1 + SZ_W)       // 270336
#define SZ_A      (ROWS * AST * 2)      // 8448: bf16 activation buffer
#define OFF_A1    (OFF_A0 + SZ_A)
#define OFF_K     (OFF_A1 + SZ_A)       // 287232
#define SZ_F      (ROWS * H_ * 4)       // 16384: f32 k buffer (tile-written, elementwise-read)
#define OFF_WD    (OFF_K + SZ_F)        // 303616
#define SZ_WD     (OUT_ * AST * 2)      // 16896: decoder W^T bf16
#define OFF_BIAS  (OFF_WD + SZ_WD)      // 4*256 f32
#define OFF_BD    (OFF_BIAS + NLAYER * H_ * 4)
#define OFF_BINIT (OFF_BD + 128)
#define OFF_TIMES (OFF_BINIT + H_ * 4)
#define SMEM_BYTES (OFF_TIMES + T_ * 4) // 326272 bytes <= 320KB/WGP

#define EPT   ((ROWS * H_) / NTHREADS)  // 16 state elements per thread (y/ksum in VGPRs)

// ---------- helpers ----------
__device__ __forceinline__ unsigned short f2bf(float f) {
  __bf16 h = (__bf16)f;                 // native v_cvt (RNE) on gfx1250
  return __builtin_bit_cast(unsigned short, h);
}

__device__ __forceinline__ float lipswish(float x) {
  return 0.909f * x * (1.0f / (1.0f + __expf(-x)));
}

union Frag { v16bf v; v8bf h[2]; };

// A fragment: 16x32 bf16 tile from activation buffer sA[m][k], row stride AST. (ISA 7.12.2)
__device__ __forceinline__ v16bf load_a_frag(const unsigned short* sA, int lane, int kt) {
  int m  = lane & 15;
  int kh = (lane >> 4) * 8;
  const unsigned short* base = sA + m * AST + kt * 32 + kh;
  Frag f;
  f.h[0] = *(const v8bf*)(const void*)(base);
  f.h[1] = *(const v8bf*)(const void*)(base + 16);
  return f.v;
}

// B fragment: 32x16 tile from transposed weights sW[n][k], row stride AST. (ISA 7.12.2)
__device__ __forceinline__ v16bf load_b_frag(const unsigned short* sW, int lane, int n0, int kbase) {
  int n  = n0 + (lane & 15);
  int kb = kbase + (lane >> 4) * 8;
  const unsigned short* base = sW + n * AST + kb;
  Frag f;
  f.h[0] = *(const v8bf*)(const void*)(base);
  f.h[1] = *(const v8bf*)(const void*)(base + 16);
  return f.v;
}

// CDNA5 async copy: one MLP layer of bf16 W^T (256x256) global -> LDS (padded rows).
// 8192 16B chunks / 256 threads = 32 async instructions per wave.
__device__ __forceinline__ void async_copy_weights(uint32_t lds_base,
                                                   const unsigned short* __restrict__ gW,
                                                   int tid) {
  #pragma unroll 4
  for (int i = 0; i < 32; ++i) {
    int c = tid + i * NTHREADS;
    int r = c >> 5;                        // weight row (n)
    int j = c & 31;                        // 16B chunk within row
    uint32_t laddr = lds_base + (uint32_t)(r * AST + j * 8) * 2u;
    uint64_t gaddr = (uint64_t)(uintptr_t)(gW + r * H_ + j * 8);
    asm volatile("global_load_async_to_lds_b128 %0, %1, off"
                 :: "v"(laddr), "v"(gaddr) : "memory");
  }
}

// ---------- prep: transpose + convert weights to bf16 in workspace ----------
__global__ void cde_prep(const float* __restrict__ Wf, const float* __restrict__ Wd,
                         unsigned short* __restrict__ WfT, unsigned short* __restrict__ WdT) {
  int i = blockIdx.x * 256 + threadIdx.x;
  if (i < NLAYER * H_ * H_) {
    int l = i >> 16, rem = i & 65535;
    int n = rem >> 8, k = rem & 255;
    WfT[i] = f2bf(Wf[l * H_ * H_ + k * H_ + n]);   // WfT[l][n][k] = Wf[l][k][n]
  } else {
    int j = i - NLAYER * H_ * H_;                  // j < OUT_*H_
    int n = j >> 8, k = j & 255;
    WdT[j] = f2bf(Wd[k * OUT_ + n]);               // WdT[n][k] = Wd[k][n]
  }
}

// ---------- main: persistent per-workgroup RK4 integration ----------
__global__ void __launch_bounds__(NTHREADS)
cde_main(const float* __restrict__ coeffs, const float* __restrict__ times,
         const float* __restrict__ W_init, const float* __restrict__ b_init,
         const float* __restrict__ bias_f, const float* __restrict__ bias_d,
         const unsigned short* __restrict__ WfT, const unsigned short* __restrict__ WdT,
         float* __restrict__ out) {
  extern __shared__ char smem[];
  unsigned short* sW0  = (unsigned short*)(smem + OFF_W0);
  unsigned short* sW1  = (unsigned short*)(smem + OFF_W1);
  unsigned short* sA0  = (unsigned short*)(smem + OFF_A0);
  unsigned short* sA1  = (unsigned short*)(smem + OFF_A1);
  float* sK    = (float*)(smem + OFF_K);
  unsigned short* sWd  = (unsigned short*)(smem + OFF_WD);
  float* sBias = (float*)(smem + OFF_BIAS);
  float* sBd   = (float*)(smem + OFF_BD);
  float* sBi   = (float*)(smem + OFF_BINIT);
  float* sT    = (float*)(smem + OFF_TIMES);

  const int tid  = threadIdx.x;
  const int lane = tid & 31;
  const int wave = tid >> 5;
  const int b0   = blockIdx.x * ROWS;
  const uint32_t ldsW0 = (uint32_t)(uintptr_t)sW0;  // LDS aperture: low 32 bits = offset
  const uint32_t ldsW1 = (uint32_t)(uintptr_t)sW1;
  const int n0w  = wave * 32;                       // 2 N-tiles of 16 per wave
  const int mb   = (lane >> 4) * 8;                 // C/D row base for this lane-half

  // per-thread RK4 state (elementwise mapping idx = tid + i*NTHREADS)
  float yreg[EPT];
  float sreg[EPT];

  // kick off W0 stream into buf0 immediately (overlaps with all the init work below)
  async_copy_weights(ldsW0, WfT, tid);

  // ---- init: constants / decoder weights into LDS; W_init^T into buf1 ----
  for (int i = tid; i < NLAYER * H_; i += NTHREADS) sBias[i] = bias_f[i];
  for (int i = tid; i < OUT_;        i += NTHREADS) sBd[i]   = bias_d[i];
  for (int i = tid; i < H_;          i += NTHREADS) sBi[i]   = b_init[i];
  for (int i = tid; i < T_;          i += NTHREADS) sT[i]    = times[i];
  for (int i = tid; i < OUT_ * H_;   i += NTHREADS) {
    int n = i >> 8, k = i & 255;
    sWd[n * AST + k] = WdT[i];
  }
  for (int i = tid; i < H_ * IN_; i += NTHREADS) {      // W_init^T (bf16) into buf1
    int n = i >> 5, k = i & 31;
    sW1[n * AST + k] = f2bf(W_init[k * H_ + n]);
  }
  for (int i = tid; i < ROWS * IN_; i += NTHREADS) {    // coeffs[:,0,:] -> bf16 A
    int m = i >> 5, k = i & 31;
    sA0[m * AST + k] = f2bf(coeffs[(size_t)(b0 + m) * T_ * IN_ + k]);
  }
  __syncthreads();

  // ---- y0 = coeffs @ W_init + b_init (K = 32, one WMMA per tile, weights from buf1) ----
  {
    v8f acc[2];
    #pragma unroll
    for (int j = 0; j < 2; ++j) {
      float b = sBi[n0w + j * 16 + (lane & 15)];
      #pragma unroll
      for (int r = 0; r < 8; ++r) acc[j][r] = b;
    }
    v16bf a = load_a_frag(sA0, lane, 0);
    #pragma unroll
    for (int j = 0; j < 2; ++j) {
      v16bf bf_ = load_b_frag(sW1, lane, n0w + j * 16, 0);
      acc[j] = __builtin_amdgcn_wmma_f32_16x16x32_bf16(false, a, false, bf_,
                                                       (short)0, acc[j], false, false);
    }
    __syncthreads();   // coeffs reads done before sA0 is overwritten
    #pragma unroll
    for (int j = 0; j < 2; ++j) {
      int ncol = n0w + j * 16 + (lane & 15);
      #pragma unroll
      for (int r = 0; r < 8; ++r) {
        float v = acc[j][r];
        sK[(mb + r) * H_ + ncol]   = v;        // stage y0 so owners can pick it up
        sA0[(mb + r) * AST + ncol] = f2bf(v);
      }
    }
  }
  __syncthreads();
  #pragma unroll
  for (int i = 0; i < EPT; ++i) yreg[i] = sK[tid + i * NTHREADS];

  // ---- time loop: decode t, then one RK4 step to t+1 ----
  for (int t = 0; t < T_; ++t) {
    // decoder: out[b, t, :] = y @ Wd + bd  (2 tiles, waves 0..1)
    if (wave < 2) {
      int n0 = wave * 16;
      v8f acc;
      float b = sBd[n0 + (lane & 15)];
      #pragma unroll
      for (int r = 0; r < 8; ++r) acc[r] = b;
      #pragma unroll
      for (int kt = 0; kt < 8; ++kt) {
        v16bf a   = load_a_frag(sA0, lane, kt);
        v16bf bf_ = load_b_frag(sWd, lane, n0, kt * 32);
        acc = __builtin_amdgcn_wmma_f32_16x16x32_bf16(false, a, false, bf_,
                                                      (short)0, acc, false, false);
      }
      int ncol = n0 + (lane & 15);
      #pragma unroll
      for (int r = 0; r < 8; ++r)
        out[((size_t)(b0 + mb + r) * T_ + t) * OUT_ + ncol] = acc[r];
    }
    if (t == T_ - 1) break;

    float dt = sT[t + 1] - sT[t];

    for (int e = 0; e < 4; ++e) {
      // f(y_in): 4 layers; weights software-pipelined through buf0/buf1 (l & 1)
      for (int l = 0; l < NLAYER; ++l) {
        __syncthreads();   // all waves done reading buf[(l+1)&1] (layer l-1) and sA buffers
        // prefetch next layer's weights into the buffer layer l-1 just released
        {
          int lnext = (l + 1) & (NLAYER - 1);
          async_copy_weights(((l + 1) & 1) ? ldsW1 : ldsW0, WfT + lnext * H_ * H_, tid);
        }
        // W(l) was issued one stage ago; in-order completion => <=32 outstanding means done
        asm volatile("s_wait_asynccnt 32" ::: "memory");
        __syncthreads();   // every wave verified its own share of W(l)

        const unsigned short* sWl  = (l & 1) ? sW1 : sW0;
        const unsigned short* sAin = (l & 1) ? sA1 : sA0;
        unsigned short*       sAon = (l & 1) ? sA0 : sA1;

        v8f acc[2];
        #pragma unroll
        for (int j = 0; j < 2; ++j) {
          float b = sBias[l * H_ + n0w + j * 16 + (lane & 15)];
          #pragma unroll
          for (int r = 0; r < 8; ++r) acc[j][r] = b;
        }
        #pragma unroll
        for (int kt = 0; kt < 8; ++kt) {
          v16bf a = load_a_frag(sAin, lane, kt);
          #pragma unroll
          for (int j = 0; j < 2; ++j) {
            v16bf bf_ = load_b_frag(sWl, lane, n0w + j * 16, kt * 32);
            acc[j] = __builtin_amdgcn_wmma_f32_16x16x32_bf16(false, a, false, bf_,
                                                             (short)0, acc[j], false, false);
          }
        }
        #pragma unroll
        for (int j = 0; j < 2; ++j) {
          int ncol = n0w + j * 16 + (lane & 15);
          if (l < NLAYER - 1) {
            #pragma unroll
            for (int r = 0; r < 8; ++r)
              sAon[(mb + r) * AST + ncol] = f2bf(lipswish(acc[j][r]));
          } else {
            #pragma unroll
            for (int r = 0; r < 8; ++r)
              sK[(mb + r) * H_ + ncol] = acc[j][r];
          }
        }
      }
      __syncthreads();   // sK complete

      // RK4 combine on per-thread registers; publish next eval's bf16 input to sA0
      #pragma unroll
      for (int i = 0; i < EPT; ++i) {
        int idx = tid + i * NTHREADS;
        int m = idx >> 8, n = idx & 255;
        float k = sK[idx];
        float a;
        if (e == 0)      { sreg[i] = k;             a = yreg[i] + 0.5f * dt * k; }
        else if (e == 1) { sreg[i] += 2.0f * k;     a = yreg[i] + 0.5f * dt * k; }
        else if (e == 2) { sreg[i] += 2.0f * k;     a = yreg[i] + dt * k;        }
        else             { yreg[i] += (dt * (1.0f / 6.0f)) * (sreg[i] + k);
                           a = yreg[i];                                          }
        sA0[m * AST + n] = f2bf(a);
      }
      __syncthreads();   // sA0 ready (next eval / decoder)
    }
  }
}

extern "C" void kernel_launch(void* const* d_in, const int* in_sizes, int n_in,
                              void* d_out, int out_size, void* d_ws, size_t ws_size,
                              hipStream_t stream) {
  (void)in_sizes; (void)n_in; (void)out_size; (void)ws_size;
  const float* coeffs = (const float*)d_in[0];
  const float* times  = (const float*)d_in[1];
  const float* W_init = (const float*)d_in[2];
  const float* b_init = (const float*)d_in[3];
  const float* Wf     = (const float*)d_in[4];
  const float* bias_f = (const float*)d_in[5];
  const float* Wd     = (const float*)d_in[6];
  const float* bias_d = (const float*)d_in[7];
  float* out = (float*)d_out;

  unsigned short* WfT = (unsigned short*)d_ws;             // 4*256*256 bf16
  unsigned short* WdT = WfT + NLAYER * H_ * H_;            // 32*256 bf16

  int prep_elems = NLAYER * H_ * H_ + OUT_ * H_;           // 270336
  cde_prep<<<dim3(prep_elems / 256), dim3(256), 0, stream>>>(Wf, Wd, WfT, WdT);

  (void)hipFuncSetAttribute(reinterpret_cast<const void*>(cde_main),
                            hipFuncAttributeMaxDynamicSharedMemorySize, SMEM_BYTES);
  cde_main<<<dim3(B_ / ROWS), dim3(NTHREADS), SMEM_BYTES, stream>>>(
      coeffs, times, W_init, b_init, bias_f, bias_d, WfT, WdT, out);
}
